// CausalSelfAttention_74423193305588
// MI455X (gfx1250) — matrix operand order repaired and checked by reading, not compiled, not run on hardware
//
#include <hip/hip_runtime.h>
#include <hip/hip_bf16.h>

typedef unsigned short u16;
typedef unsigned int u32;
typedef __attribute__((ext_vector_type(16))) __bf16 v16bf;
typedef __attribute__((ext_vector_type(8))) float v8f;

// pointee/address-space types matching the async-LDS builtin's signature
typedef int gv4i __attribute__((vector_size(16)));
typedef __attribute__((address_space(1))) gv4i gv4i_g;  // global
typedef __attribute__((address_space(3))) gv4i gv4i_l;  // LDS

#define T_SEQ 2048
#define EMB 2048
#define NHEAD 16
#define HDIM 128
#define VSTRIDE 72

union FragU {
  uint4 q[2];
  v16bf v;
};

__device__ inline u16 f2bf(float f) {
  u32 u = __builtin_bit_cast(u32, f);
  u += 0x7fffu + ((u >> 16) & 1u);   // round-to-nearest-even
  return (u16)(u >> 16);
}

__device__ inline v8f vzero8() {
  v8f v;
#pragma unroll
  for (int i = 0; i < 8; ++i) v[i] = 0.0f;
  return v;
}

__device__ inline v8f wmma_bf16(v16bf a, v16bf b, v8f c) {
  return __builtin_amdgcn_wmma_f32_16x16x32_bf16(false, a, false, b, (short)0, c,
                                                 false, false);
}

// ---------------------------------------------------------------------------
// Async global -> LDS 128-bit copy (CDNA5 GLOBAL_LOAD_ASYNC_TO_LDS_B128,
// tracked by ASYNCcnt). Builtin if present, inline asm on the device pass
// otherwise, plain copy for the host pass.
// ---------------------------------------------------------------------------
__device__ inline void async_copy_b128(const u16* g, u16* l) {
#if __has_builtin(__builtin_amdgcn_global_load_async_to_lds_b128)
  __builtin_amdgcn_global_load_async_to_lds_b128(
      (gv4i_g*)(u16*)g, (gv4i_l*)l, 0, 0);
#elif defined(__AMDGCN__)
  u32 la = (u32)(size_t)(__attribute__((address_space(3))) u16*)l;
  unsigned long long ga = (unsigned long long)(size_t)g;
  asm volatile("global_load_async_to_lds_b128 %0, %1, off"
               :
               : "v"(la), "v"(ga)
               : "memory");
#else
  *(uint4*)l = *(const uint4*)g;
#endif
}

__device__ inline void wait_async() {
#if __has_builtin(__builtin_amdgcn_s_wait_asynccnt)
  __builtin_amdgcn_s_wait_asynccnt(0);
#elif defined(__AMDGCN__)
  asm volatile("s_wait_asynccnt 0" ::: "memory");
#endif
}

// ---------------------------------------------------------------------------
// fp32 -> bf16 cast
// ---------------------------------------------------------------------------
__global__ void cast_bf16_kernel(const float* __restrict__ in,
                                 u16* __restrict__ out, int n) {
  int i = blockIdx.x * blockDim.x + threadIdx.x;
  if (i < n) out[i] = f2bf(in[i]);
}

// ---------------------------------------------------------------------------
// C[M,N] (f32) = A[M,K] (bf16, row-major) * B[N,K]^T (bf16, row-major rows=N)
// 128x128 block tile, BK=32, 8 waves, each wave 64x32 via 4x2 WMMA tiles.
// Tiles staged via async global->LDS copies.
// ---------------------------------------------------------------------------
__global__ __launch_bounds__(256) void gemm_bf16_kernel(
    const u16* __restrict__ A, const u16* __restrict__ B, float* __restrict__ C,
    int M, int N, int K) {
  __shared__ __align__(16) u16 As[128 * 32];
  __shared__ __align__(16) u16 Bs[128 * 32];

  const int tid = threadIdx.x;
  const int lane = tid & 31;
  const int wid = tid >> 5;
  const int hl = lane >> 4;   // lane half (0/1)
  const int l16 = lane & 15;
  const int bm = blockIdx.y * 128;
  const int bn = blockIdx.x * 128;
  const int wm = (wid >> 2) * 64;   // wave M offset within block
  const int wn = (wid & 3) * 32;    // wave N offset within block

  v8f acc[4][2];
#pragma unroll
  for (int mt = 0; mt < 4; ++mt)
#pragma unroll
    for (int nt = 0; nt < 2; ++nt) acc[mt][nt] = vzero8();

  for (int k0 = 0; k0 < K; k0 += 32) {
    // cooperative tile stage: 128 rows x 32 bf16 = 512 x b128 per tile,
    // async DMA straight into LDS (no VGPR round trip)
#pragma unroll
    for (int i = 0; i < 2; ++i) {
      int idx = tid + i * 256;
      int row = idx >> 2;
      int c8 = (idx & 3) * 8;
      async_copy_b128(&A[(size_t)(bm + row) * K + k0 + c8], &As[row * 32 + c8]);
      async_copy_b128(&B[(size_t)(bn + row) * K + k0 + c8], &Bs[row * 32 + c8]);
    }
    wait_async();
    __syncthreads();

    v16bf af[4], bfr[2];
#pragma unroll
    for (int mt = 0; mt < 4; ++mt) {
      // A fragment: lane = row, K chunks {0..7,16..23} / {8..15,24..31}
      const u16* ap = &As[(wm + mt * 16 + l16) * 32 + hl * 8];
      FragU f;
      f.q[0] = *(const uint4*)ap;
      f.q[1] = *(const uint4*)(ap + 16);
      af[mt] = f.v;
    }
#pragma unroll
    for (int nt = 0; nt < 2; ++nt) {
      // B fragment: lane = column, K chunks {0..7,16..23} / {8..15,24..31}
      const u16* bp = &Bs[(wn + nt * 16 + l16) * 32 + hl * 8];
      FragU f;
      f.q[0] = *(const uint4*)bp;
      f.q[1] = *(const uint4*)(bp + 16);
      bfr[nt] = f.v;
    }
#pragma unroll
    for (int mt = 0; mt < 4; ++mt)
#pragma unroll
      for (int nt = 0; nt < 2; ++nt)
        acc[mt][nt] = wmma_bf16(af[mt], bfr[nt], acc[mt][nt]);
    __syncthreads();
  }

  // C layout: lane = N column (l16), VGPR r = row r (+8 for upper lane half)
#pragma unroll
  for (int mt = 0; mt < 4; ++mt) {
#pragma unroll
    for (int nt = 0; nt < 2; ++nt) {
      int row0 = bm + wm + mt * 16 + hl * 8;
      int col = bn + wn + nt * 16 + l16;
#pragma unroll
      for (int r = 0; r < 8; ++r)
        C[(size_t)(row0 + r) * N + col] = acc[mt][nt][r];
    }
  }
}

// ---------------------------------------------------------------------------
// RoPE on Q/K + head-major bf16 split of qkv. grid=(T, H), block=64 (pairs).
// ---------------------------------------------------------------------------
__global__ void rope_split_kernel(const float* __restrict__ qkv,
                                  u16* __restrict__ qh, u16* __restrict__ kh,
                                  u16* __restrict__ vh) {
  const int t = blockIdx.x;
  const int h = blockIdx.y;
  const int i = threadIdx.x;  // pair index 0..63
  const float theta =
      __expf(-((float)(2 * i) / 128.0f) * 9.210340371976184f);  // ln(10000)
  const float ang = (float)t * theta;
  float sn, cs;
  __sincosf(ang, &sn, &cs);
  const int qb = t * 3 * EMB + h * HDIM;
  const int kb = qb + EMB;
  const int vb = qb + 2 * EMB;
  const int ob = (h * T_SEQ + t) * HDIM;

  float q0 = qkv[qb + 2 * i], q1 = qkv[qb + 2 * i + 1];
  qh[ob + 2 * i] = f2bf(q0 * cs - q1 * sn);
  qh[ob + 2 * i + 1] = f2bf(q1 * cs + q0 * sn);
  float k0 = qkv[kb + 2 * i], k1 = qkv[kb + 2 * i + 1];
  kh[ob + 2 * i] = f2bf(k0 * cs - k1 * sn);
  kh[ob + 2 * i + 1] = f2bf(k1 * cs + k0 * sn);
  vh[ob + 2 * i] = f2bf(qkv[vb + 2 * i]);
  vh[ob + 2 * i + 1] = f2bf(qkv[vb + 2 * i + 1]);
}

// ---------------------------------------------------------------------------
// Flash attention, causal. grid=(T/64, H), 128 threads = 4 waves,
// each wave owns 16 q-rows. K tile staged via async DMA, V tile transposed
// in LDS, online softmax with half-wave shuffles.
// ---------------------------------------------------------------------------
__global__ __launch_bounds__(128) void flash_attn_kernel(
    const u16* __restrict__ qh, const u16* __restrict__ kh,
    const u16* __restrict__ vh, u16* __restrict__ oh) {
  __shared__ __align__(16) u16 Ks[64 * 128];
  __shared__ __align__(16) u16 Vs[128 * VSTRIDE];  // transposed: [d][key]
  __shared__ __align__(16) u16 Ps[4][16 * 64];

  const int tid = threadIdx.x;
  const int lane = tid & 31;
  const int wid = tid >> 5;
  const int hl = lane >> 4;
  const int l16 = lane & 15;
  const int head = blockIdx.y;
  const int qt0 = blockIdx.x * 64;
  const int qrow0 = qt0 + wid * 16;
  const size_t hbase = (size_t)head * T_SEQ * HDIM;

  // Q fragments resident in VGPRs (16 rows x 128 d = 4 A-fragments of K=32)
  v16bf qf[4];
#pragma unroll
  for (int kc = 0; kc < 4; ++kc) {
    const u16* qp = &qh[hbase + (size_t)(qrow0 + l16) * HDIM + kc * 32 + hl * 8];
    FragU f;
    f.q[0] = *(const uint4*)qp;
    f.q[1] = *(const uint4*)(qp + 16);
    qf[kc] = f.v;
  }

  v8f o[8];
#pragma unroll
  for (int dt = 0; dt < 8; ++dt) o[dt] = vzero8();
  float rowm[8], rowl[8], rowa[8];
#pragma unroll
  for (int r = 0; r < 8; ++r) {
    rowm[r] = -1e30f;
    rowl[r] = 0.0f;
  }

  const float scale = 0.08838834764831845f;  // 1/sqrt(128)
  const int nkt = blockIdx.x + 1;            // causal: only tiles up to diagonal
  for (int kt = 0; kt < nkt; ++kt) {
    const int kt0 = kt * 64;
    __syncthreads();  // protect K/V tiles from previous iteration's readers
    // K tile [64 x 128] row-major via async DMA; V tile transposed [128 x 64]
#pragma unroll
    for (int i = 0; i < 8; ++i) {
      int idx = tid + i * 128;
      int row = idx >> 4;
      int c8 = (idx & 15) * 8;
      async_copy_b128(&kh[hbase + (size_t)(kt0 + row) * HDIM + c8],
                      &Ks[row * 128 + c8]);
      uint4 vv = *(const uint4*)&vh[hbase + (size_t)(kt0 + row) * HDIM + c8];
      const u16* ve = (const u16*)&vv;
#pragma unroll
      for (int j = 0; j < 8; ++j) Vs[(c8 + j) * VSTRIDE + row] = ve[j];
    }
    wait_async();
    __syncthreads();

    // S = Q K^T : 4 tiles of 16 keys, 4 K-steps of 32 each
    v8f s[4];
#pragma unroll
    for (int nt = 0; nt < 4; ++nt) {
      v8f a = vzero8();
#pragma unroll
      for (int kc = 0; kc < 4; ++kc) {
        const u16* kp = &Ks[(nt * 16 + l16) * 128 + kc * 32 + hl * 8];
        FragU f;
        f.q[0] = *(const uint4*)kp;
        f.q[1] = *(const uint4*)(kp + 16);
        a = wmma_bf16(qf[kc], f.v, a);
      }
      s[nt] = a;
    }

    // scale + causal mask (element (r, l16) = (row hl*8+r, key nt*16+l16))
#pragma unroll
    for (int nt = 0; nt < 4; ++nt) {
      int kpos = kt0 + nt * 16 + l16;
#pragma unroll
      for (int r = 0; r < 8; ++r) {
        int qpos = qrow0 + hl * 8 + r;
        float v = s[nt][r] * scale;
        s[nt][r] = (kpos <= qpos) ? v : -1e30f;
      }
    }

    // online softmax (row reductions across the 16-lane half-wave)
#pragma unroll
    for (int r = 0; r < 8; ++r) {
      float mx = s[0][r];
#pragma unroll
      for (int nt = 1; nt < 4; ++nt) mx = fmaxf(mx, s[nt][r]);
#pragma unroll
      for (int off2 = 1; off2 < 16; off2 <<= 1)
        mx = fmaxf(mx, __shfl_xor(mx, off2, 32));
      float newm = fmaxf(rowm[r], mx);
      float alpha = __expf(rowm[r] - newm);
      float sm = 0.0f;
#pragma unroll
      for (int nt = 0; nt < 4; ++nt) {
        float p = __expf(s[nt][r] - newm);
        s[nt][r] = p;
        sm += p;
      }
#pragma unroll
      for (int off2 = 1; off2 < 16; off2 <<= 1) sm += __shfl_xor(sm, off2, 32);
      rowl[r] = rowl[r] * alpha + sm;
      rowm[r] = newm;
      rowa[r] = alpha;
    }

    // rescale accumulators
#pragma unroll
    for (int dt = 0; dt < 8; ++dt)
#pragma unroll
      for (int r = 0; r < 8; ++r) o[dt][r] *= rowa[r];

    // C-layout -> A-layout for P via per-wave LDS tile (bf16)
#pragma unroll
    for (int nt = 0; nt < 4; ++nt)
#pragma unroll
      for (int r = 0; r < 8; ++r)
        Ps[wid][(hl * 8 + r) * 64 + nt * 16 + l16] = f2bf(s[nt][r]);
    __syncthreads();

    // O += P V : 2 K-steps of 32 keys, 8 d-tiles
#pragma unroll
    for (int kc2 = 0; kc2 < 2; ++kc2) {
      const u16* pp = &Ps[wid][l16 * 64 + kc2 * 32 + hl * 8];
      FragU fp;
      fp.q[0] = *(const uint4*)pp;
      fp.q[1] = *(const uint4*)(pp + 16);
#pragma unroll
      for (int dt = 0; dt < 8; ++dt) {
        const u16* vp = &Vs[(dt * 16 + l16) * VSTRIDE + kc2 * 32 + hl * 8];
        FragU fv;
        fv.q[0] = *(const uint4*)vp;
        fv.q[1] = *(const uint4*)(vp + 16);
        o[dt] = wmma_bf16(fp.v, fv.v, o[dt]);
      }
    }
  }

  // normalize and store O as bf16 in [t][E] layout for the output projection
#pragma unroll
  for (int r = 0; r < 8; ++r) rowa[r] = 1.0f / rowl[r];
#pragma unroll
  for (int dt = 0; dt < 8; ++dt) {
    int col = head * HDIM + dt * 16 + l16;
#pragma unroll
    for (int r = 0; r < 8; ++r) {
      int row = qrow0 + hl * 8 + r;
      oh[(size_t)row * EMB + col] = f2bf(o[dt][r] * rowa[r]);
    }
  }
}

// ---------------------------------------------------------------------------
extern "C" void kernel_launch(void* const* d_in, const int* in_sizes, int n_in,
                              void* d_out, int out_size, void* d_ws,
                              size_t ws_size, hipStream_t stream) {
  (void)in_sizes;
  (void)n_in;
  (void)out_size;
  (void)ws_size;
  const float* x = (const float*)d_in[0];
  const float* w_attn = (const float*)d_in[1];
  const float* w_proj = (const float*)d_in[2];
  float* out = (float*)d_out;

  char* ws = (char*)d_ws;
  size_t off = 0;
  auto alloc = [&](size_t bytes) -> void* {
    void* p = ws + off;
    off += (bytes + 255) & ~(size_t)255;
    return p;
  };

  const size_t nX = (size_t)T_SEQ * EMB;       // 4,194,304
  const size_t nWA = (size_t)3 * EMB * EMB;    // 12,582,912
  const size_t nWP = (size_t)EMB * EMB;
  const size_t nH = (size_t)NHEAD * T_SEQ * HDIM;

  u16* xb = (u16*)alloc(nX * 2);
  u16* wab = (u16*)alloc(nWA * 2);
  u16* wpb = (u16*)alloc(nWP * 2);
  float* qkv = (float*)alloc((size_t)T_SEQ * 3 * EMB * 4);
  u16* qhb = (u16*)alloc(nH * 2);
  u16* khb = (u16*)alloc(nH * 2);
  u16* vhb = (u16*)alloc(nH * 2);
  u16* ohb = (u16*)alloc(nX * 2);

  cast_bf16_kernel<<<(int)((nX + 255) / 256), 256, 0, stream>>>(x, xb, (int)nX);
  cast_bf16_kernel<<<(int)((nWA + 255) / 256), 256, 0, stream>>>(w_attn, wab,
                                                                 (int)nWA);
  cast_bf16_kernel<<<(int)((nWP + 255) / 256), 256, 0, stream>>>(w_proj, wpb,
                                                                 (int)nWP);

  // qkv = x @ w_attn^T
  gemm_bf16_kernel<<<dim3(3 * EMB / 128, T_SEQ / 128), 256, 0, stream>>>(
      xb, wab, qkv, T_SEQ, 3 * EMB, EMB);
  // RoPE + head-major split to bf16
  rope_split_kernel<<<dim3(T_SEQ, NHEAD), 64, 0, stream>>>(qkv, qhb, khb, vhb);
  // causal flash attention
  flash_attn_kernel<<<dim3(T_SEQ / 64, NHEAD), 128, 0, stream>>>(qhb, khb, vhb,
                                                                 ohb);
  // out = O @ w_proj^T
  gemm_bf16_kernel<<<dim3(EMB / 128, T_SEQ / 128), 256, 0, stream>>>(
      ohb, wpb, out, T_SEQ, EMB, EMB);
}
